// LanguagePlanningHead_79585743995165
// MI455X (gfx1250) — matrix operand order repaired and checked by reading, not compiled, not run on hardware
//
#include <hip/hip_runtime.h>
#include <hip/hip_bf16.h>
#include <math.h>

#define L_ 12
#define B_ 4
#define N_ 512
#define S_ 512
#define V_ 32000
#define H_ 768
#define HEADS_ 8
#define DH_ 96
#define LTOT_ (S_ + N_)

typedef __attribute__((ext_vector_type(16))) _Float16 v16h;
typedef __attribute__((ext_vector_type(8)))  _Float16 v8h;
typedef __attribute__((ext_vector_type(4)))  _Float16 v4h;
typedef __attribute__((ext_vector_type(8)))  float    v8f;

// ---------------------------------------------------------------------------
// Generic batched GEMM:  C[z] = act( alpha * A[z] @ B[z] + bias + R[z] )
// A: [M,K] row-major f32.  B: [K,N] row-major, or [N,K] if bTrans (f32).
// REQUIREMENTS (all call sites satisfy them): M % 128 == 0, K % 32 == 0,
// Ncols % 4 == 0, all row pitches % 4 == 0 (16B-aligned float4 loads).
// f32 -> f16 conversion during LDS staging; math via v_wmma_f32_16x16x32_f16.
// Block = 256 threads = 8 wave32 waves; tile 128(M) x 64(N) x 32(K),
// double-buffered LDS; each wave owns a 32x32 tile = 2x2 WMMA accumulators.
// LDS layouts are WMMA-fragment-major: each lane's fragment is two contiguous
// 16B chunks -> ds_load_b128 pairs.
// ---------------------------------------------------------------------------
__global__ __launch_bounds__(256) void wmma_gemm_kernel(
    const float* __restrict__ A, long ldA, long sAz,
    const float* __restrict__ B, long ldB, long sBz, int bTrans,
    float* __restrict__ C, long ldC, long sC1, long sC2, int cDiv,
    const float* __restrict__ bias,
    const float* __restrict__ R,
    int M, int Ncols, int K, float alpha, int act)
{
    __shared__ _Float16 As[2][128][40];   // 40-half pitch: 80B, 16B aligned
    __shared__ _Float16 Bs[2][64][40];    // stored transposed: [col][k]

    const int t = threadIdx.x;
    const int z = blockIdx.z;
    const long zoff = (long)(z / cDiv) * sC1 + (long)(z % cDiv) * sC2;
    const float* Ab = A + (long)z * sAz;
    const float* Bb = B + (long)z * sBz;
    float* Cb = C + zoff;
    const float* Rb = R ? (R + zoff) : (const float*)0;

    const int row0 = blockIdx.y * 128;
    const int col0 = blockIdx.x * 64;

    const int w    = t >> 5;       // wave 0..7
    const int lane = t & 31;
    const int wm   = w >> 1;       // 0..3 -> 32-row strip
    const int wn   = w & 1;        // 0..1 -> 32-col strip
    const int n16  = lane & 15;
    const int half = lane >> 4;

    v8f acc[2][2] = {};
    float4 aR[4];                  // staged A: 4 x float4 per thread
    float4 bR[2];                  // staged B: 2 x float4 per thread

    // ---- global loads (unconditional, b128) ----
    auto gload = [&](int k0) {
        #pragma unroll
        for (int i = 0; i < 4; ++i) {               // A: 128x32 = 1024 float4
            int e = i * 256 + t;
            int r = e >> 3, c4 = e & 7;             // 8 float4 per row
            aR[i] = *(const float4*)(Ab + (long)(row0 + r) * ldA + k0 + c4 * 4);
        }
        if (!bTrans) {
            #pragma unroll
            for (int i = 0; i < 2; ++i) {           // B: 32x64 = 512 float4
                int e = i * 256 + t;
                int r = e >> 4, c4 = e & 15;        // along N
                int gc = col0 + c4 * 4;
                int gcl = (gc < Ncols) ? gc : (Ncols - 4);   // clamp, select 0 later
                float4 v = *(const float4*)(Bb + (long)(k0 + r) * ldB + gcl);
                if (gc >= Ncols) { v.x = 0.f; v.y = 0.f; v.z = 0.f; v.w = 0.f; }
                bR[i] = v;
            }
        } else {
            #pragma unroll
            for (int i = 0; i < 2; ++i) {
                int e = i * 256 + t;
                int c = e >> 3, r4 = e & 7;         // along K (contiguous)
                int gc = col0 + c;
                int gcl = (gc < Ncols) ? gc : (Ncols - 1);
                float4 v = *(const float4*)(Bb + (long)gcl * ldB + k0 + r4 * 4);
                if (gc >= Ncols) { v.x = 0.f; v.y = 0.f; v.z = 0.f; v.w = 0.f; }
                bR[i] = v;
            }
        }
    };

    // ---- convert + store staged registers into LDS buffer ----
    auto stage = [&](int buf) {
        #pragma unroll
        for (int i = 0; i < 4; ++i) {
            int e = i * 256 + t;
            int r = e >> 3, c4 = e & 7;
            v4h h;
            h[0] = (_Float16)aR[i].x; h[1] = (_Float16)aR[i].y;
            h[2] = (_Float16)aR[i].z; h[3] = (_Float16)aR[i].w;
            *(v4h*)&As[buf][r][c4 * 4] = h;
        }
        if (!bTrans) {
            #pragma unroll
            for (int i = 0; i < 2; ++i) {
                int e = i * 256 + t;
                int r = e >> 4, c4 = e & 15;
                Bs[buf][c4 * 4 + 0][r] = (_Float16)bR[i].x;
                Bs[buf][c4 * 4 + 1][r] = (_Float16)bR[i].y;
                Bs[buf][c4 * 4 + 2][r] = (_Float16)bR[i].z;
                Bs[buf][c4 * 4 + 3][r] = (_Float16)bR[i].w;
            }
        } else {
            #pragma unroll
            for (int i = 0; i < 2; ++i) {
                int e = i * 256 + t;
                int c = e >> 3, r4 = e & 7;
                v4h h;
                h[0] = (_Float16)bR[i].x; h[1] = (_Float16)bR[i].y;
                h[2] = (_Float16)bR[i].z; h[3] = (_Float16)bR[i].w;
                *(v4h*)&Bs[buf][c][r4 * 4] = h;
            }
        }
    };

    // ---- WMMA compute from LDS buffer: fragments via ds_load_b128 pairs ----
    auto compute = [&](int buf) {
        v16h afr[2], bfr[2];
        #pragma unroll
        for (int ti = 0; ti < 2; ++ti) {
            int row = wm * 32 + ti * 16 + n16;      // ISA A layout: row = lane&15
            v8h lo = *(const v8h*)&As[buf][row][half * 8];        // K 0-7 / 8-15
            v8h hi = *(const v8h*)&As[buf][row][16 + half * 8];   // K 16-23 / 24-31
            afr[ti] = __builtin_shufflevector(lo, hi,
                        0,1,2,3,4,5,6,7,8,9,10,11,12,13,14,15);
        }
        #pragma unroll
        for (int tj = 0; tj < 2; ++tj) {
            int col = wn * 32 + tj * 16 + n16;      // ISA B layout: col = lane&15
            v8h blo = *(const v8h*)&Bs[buf][col][half * 8];        // K 0-7 / 8-15
            v8h bhi = *(const v8h*)&Bs[buf][col][16 + half * 8];   // K 16-23 / 24-31
            bfr[tj] = __builtin_shufflevector(blo, bhi,
                        0,1,2,3,4,5,6,7,8,9,10,11,12,13,14,15);
        }
        acc[0][0] = __builtin_amdgcn_wmma_f32_16x16x32_f16(false, afr[0], false, bfr[0], (short)0, acc[0][0], false, false);
        acc[0][1] = __builtin_amdgcn_wmma_f32_16x16x32_f16(false, afr[0], false, bfr[1], (short)0, acc[0][1], false, false);
        acc[1][0] = __builtin_amdgcn_wmma_f32_16x16x32_f16(false, afr[1], false, bfr[0], (short)0, acc[1][0], false, false);
        acc[1][1] = __builtin_amdgcn_wmma_f32_16x16x32_f16(false, afr[1], false, bfr[1], (short)0, acc[1][1], false, false);
    };

    // ---- double-buffered main loop ----
    gload(0);
    stage(0);
    __syncthreads();
    int buf = 0;
    for (int k0 = 0; k0 < K; k0 += 32) {
        const bool more = (k0 + 32) < K;
        if (more) gload(k0 + 32);      // global b128 loads overlap WMMA below
        compute(buf);
        if (more) stage(buf ^ 1);      // write other buffer
        __syncthreads();
        buf ^= 1;
    }

    // ---- epilogue: alpha, bias, residual, activation, store ----
    #pragma unroll
    for (int ti = 0; ti < 2; ++ti)
        #pragma unroll
        for (int tj = 0; tj < 2; ++tj)
            #pragma unroll
            for (int r = 0; r < 8; ++r) {
                int gm = row0 + wm * 32 + ti * 16 + half * 8 + r;
                int gn = col0 + wn * 32 + tj * 16 + n16;
                if (gn < Ncols) {
                    float v = acc[ti][tj][r] * alpha;
                    if (bias) v += bias[gn];
                    if (Rb)   v += Rb[(long)gm * ldC + gn];
                    if (act == 1)  // exact GELU
                        v = 0.5f * v * (1.0f + erff(v * 0.70710678118654752f));
                    Cb[(long)gm * ldC + gn] = v;
                }
            }
}

// ---------------------------------------------------------------------------
// Elementwise / reduction kernels
// ---------------------------------------------------------------------------
__global__ void embed_kernel(const int* __restrict__ ids,
                             const float* __restrict__ emb,
                             float* __restrict__ out, long total) {
    long i = (long)blockIdx.x * blockDim.x + threadIdx.x;
    if (i >= total) return;
    int c = (int)(i % H_);
    long tok = i / H_;
    out[i] = emb[(long)ids[tok] * H_ + c] * 27.712812921102035f;  // sqrt(768)
}

__global__ __launch_bounds__(256) void rmsnorm_kernel(const float* __restrict__ x,
                                                      const float* __restrict__ w,
                                                      float* __restrict__ out) {
    long row = blockIdx.x;
    const float* xr = x + row * H_;
    float* orow = out + row * H_;
    __shared__ float red[256];
    float s = 0.0f;
    for (int c = threadIdx.x; c < H_; c += 256) { float v = xr[c]; s += v * v; }
    red[threadIdx.x] = s; __syncthreads();
    for (int o = 128; o > 0; o >>= 1) {
        if (threadIdx.x < o) red[threadIdx.x] += red[threadIdx.x + o];
        __syncthreads();
    }
    float inv = rsqrtf(red[0] / (float)H_ + 1e-6f);
    __syncthreads();
    for (int c = threadIdx.x; c < H_; c += 256)
        orow[c] = xr[c] * inv * (1.0f + w[c]);
}

// qkv [B,N,3H] -> RoPE(q) -> q_arr[B,h,N,d]; RoPE(k)/v -> keys/vals[B,h,S+N,d]
__global__ void qkv_post_kernel(const float* __restrict__ qkv,
                                float* __restrict__ q_arr,
                                float* __restrict__ keys,
                                float* __restrict__ vals, long total) {
    long idx = (long)blockIdx.x * blockDim.x + threadIdx.x;
    if (idx >= total) return;                       // total = B*N*HEADS*48
    int d = (int)(idx % 48); long t = idx / 48;
    int h = (int)(t % HEADS_); t /= HEADS_;
    int n = (int)(t % N_);
    int b = (int)(t / N_);
    const float* base = qkv + ((long)b * N_ + n) * (3 * H_);
    int col = h * DH_;
    float q1 = base[col + d],          q2 = base[col + d + 48];
    float k1 = base[H_ + col + d],     k2 = base[H_ + col + d + 48];
    float v1 = base[2 * H_ + col + d], v2 = base[2 * H_ + col + d + 48];
    float freq = powf(10000.0f, -(float)(2 * d) / 96.0f);
    float sn, cs;
    __sincosf((float)n * freq, &sn, &cs);
    long qoff = (((long)b * HEADS_ + h) * N_ + n) * DH_;
    q_arr[qoff + d]      = q1 * cs - q2 * sn;
    q_arr[qoff + d + 48] = q2 * cs + q1 * sn;
    long koff = (((long)b * HEADS_ + h) * LTOT_ + (S_ + n)) * DH_;
    keys[koff + d]      = k1 * cs - k2 * sn;
    keys[koff + d + 48] = k2 * cs + k1 * sn;
    vals[koff + d]      = v1;
    vals[koff + d + 48] = v2;
}

// sk/sv [B,S,H] -> keys/vals[B,h,0..S-1,d]
__global__ void skv_post_kernel(const float* __restrict__ sk,
                                const float* __restrict__ sv,
                                float* __restrict__ keys,
                                float* __restrict__ vals, long total) {
    long idx = (long)blockIdx.x * blockDim.x + threadIdx.x;
    if (idx >= total) return;                       // total = B*S*H
    int c = (int)(idx % H_); long t = idx / H_;
    int j = (int)(t % S_);
    int b = (int)(t / S_);
    int h = c / DH_, d = c % DH_;
    long off = (((long)b * HEADS_ + h) * LTOT_ + j) * DH_ + d;
    keys[off] = sk[idx];
    vals[off] = sv[idx];
}

// masked softmax over rows of sim [B*HEADS*N, LTOT]; mask: j <= (row%N)+S
__global__ __launch_bounds__(256) void softmax_kernel(float* __restrict__ sim) {
    long row = blockIdx.x;
    int i = (int)(row % N_);
    float* p = sim + row * LTOT_;
    int jmax = S_ + i;
    __shared__ float red[256];
    float lmax = -1e30f;
    for (int j = threadIdx.x; j < LTOT_; j += 256)
        if (j <= jmax) lmax = fmaxf(lmax, p[j]);
    red[threadIdx.x] = lmax; __syncthreads();
    for (int o = 128; o > 0; o >>= 1) {
        if (threadIdx.x < o) red[threadIdx.x] = fmaxf(red[threadIdx.x], red[threadIdx.x + o]);
        __syncthreads();
    }
    float gmax = red[0]; __syncthreads();
    float lsum = 0.0f;
    for (int j = threadIdx.x; j < LTOT_; j += 256) {
        float e = (j <= jmax) ? __expf(p[j] - gmax) : 0.0f;
        p[j] = e; lsum += e;
    }
    red[threadIdx.x] = lsum; __syncthreads();
    for (int o = 128; o > 0; o >>= 1) {
        if (threadIdx.x < o) red[threadIdx.x] += red[threadIdx.x + o];
        __syncthreads();
    }
    float inv = 1.0f / red[0];
    __syncthreads();
    for (int j = threadIdx.x; j < LTOT_; j += 256) p[j] *= inv;
}

// ---------------------------------------------------------------------------
// Host side
// ---------------------------------------------------------------------------
static void launch_gemm(hipStream_t st,
                        const float* A, long ldA, long sAz,
                        const float* B, long ldB, long sBz, int bT,
                        float* C, long ldC, long sC1, long sC2, int cDiv,
                        const float* bias, const float* R,
                        int M, int Nc, int K, float alpha, int act, int batches) {
    dim3 grid((Nc + 63) / 64, (M + 127) / 128, batches);
    wmma_gemm_kernel<<<grid, 256, 0, st>>>(A, ldA, sAz, B, ldB, sBz, bT,
                                           C, ldC, sC1, sC2, cDiv, bias, R,
                                           M, Nc, K, alpha, act);
}

extern "C" void kernel_launch(void* const* d_in, const int* in_sizes, int n_in,
                              void* d_out, int out_size, void* d_ws, size_t ws_size,
                              hipStream_t stream) {
    const int*   ids     = (const int*)  d_in[0];
    const float* vlm_key = (const float*)d_in[1];   // [L,B,S,H]
    const float* vlm_val = (const float*)d_in[2];
    const float* embedW  = (const float*)d_in[3];   // [V,H]
    const float* n1w     = (const float*)d_in[4];
    const float* n2w     = (const float*)d_in[5];
    const float* qkvW    = (const float*)d_in[6];   // [L,H,3H]
    const float* qkvb    = (const float*)d_in[7];
    const float* skvW    = (const float*)d_in[8];   // [L,H,H]
    const float* skvb    = (const float*)d_in[9];
    const float* outW    = (const float*)d_in[10];
    const float* outb    = (const float*)d_in[11];
    const float* f1W     = (const float*)d_in[12];
    const float* f1b     = (const float*)d_in[13];
    const float* f2W     = (const float*)d_in[14];
    const float* f2b     = (const float*)d_in[15];
    float* logits = (float*)d_out;

    const long BNH   = (long)B_ * N_ * H_;            // 1,572,864
    const long BHLD  = (long)B_ * HEADS_ * LTOT_ * DH_;
    const long SIMSZ = (long)B_ * HEADS_ * N_ * LTOT_;

    float* hidden  = (float*)d_ws;
    float* xnorm   = hidden  + BNH;
    float* qkvbuf  = xnorm   + BNH;                   // [B,N,3H]
    float* q_arr   = qkvbuf  + 3 * BNH;               // [B,h,N,d]
    float* keys    = q_arr   + BNH;                   // [B,h,S+N,d]
    float* vals    = keys    + BHLD;
    float* skbuf   = vals    + BHLD;                  // [B,S,H]
    float* svbuf   = skbuf   + BNH;
    float* simbuf  = svbuf   + BNH;                   // [B*h,N,S+N]
    float* attn_o  = simbuf  + SIMSZ;                 // [B,N,H]
    float* ffmid   = attn_o  + BNH;

    const float scale = 0.10206207261596577f;         // 96^-0.5
    const int rows = B_ * N_;                         // 2048

    // hidden = embed[ids] * sqrt(H)
    embed_kernel<<<(unsigned)((BNH + 255) / 256), 256, 0, stream>>>(ids, embedW, hidden, BNH);

    for (int l = 0; l < L_; ++l) {
        // x = rmsnorm(hidden, norm1)
        rmsnorm_kernel<<<rows, 256, 0, stream>>>(hidden, n1w + (long)l * H_, xnorm);

        // qkv = x @ qkv_W + qkv_b            [2048,768]x[768,2304]
        launch_gemm(stream, xnorm, H_, 0,
                    qkvW + (long)l * H_ * 3 * H_, 3 * H_, 0, 0,
                    qkvbuf, 3 * H_, 0, 0, 1,
                    qkvb + (long)l * 3 * H_, nullptr,
                    rows, 3 * H_, H_, 1.0f, 0, 1);

        // sk = vlm_key[l] @ skv_W + b ; sv = vlm_value[l] @ skv_W + b
        launch_gemm(stream, vlm_key + (long)l * BNH, H_, 0,
                    skvW + (long)l * H_ * H_, H_, 0, 0,
                    skbuf, H_, 0, 0, 1,
                    skvb + (long)l * H_, nullptr,
                    B_ * S_, H_, H_, 1.0f, 0, 1);
        launch_gemm(stream, vlm_val + (long)l * BNH, H_, 0,
                    skvW + (long)l * H_ * H_, H_, 0, 0,
                    svbuf, H_, 0, 0, 1,
                    skvb + (long)l * H_, nullptr,
                    B_ * S_, H_, H_, 1.0f, 0, 1);

        // RoPE + scatter to per-head layouts
        {
            long tq = (long)B_ * N_ * HEADS_ * 48;
            qkv_post_kernel<<<(unsigned)((tq + 255) / 256), 256, 0, stream>>>(qkvbuf, q_arr, keys, vals, tq);
            long ts = (long)B_ * S_ * H_;
            skv_post_kernel<<<(unsigned)((ts + 255) / 256), 256, 0, stream>>>(skbuf, svbuf, keys, vals, ts);
        }

        // sim = scale * q @ keys^T           batched over 32 (b,h)
        launch_gemm(stream, q_arr, DH_, (long)N_ * DH_,
                    keys, DH_, (long)LTOT_ * DH_, 1,
                    simbuf, LTOT_, (long)N_ * LTOT_, 0, 1,
                    nullptr, nullptr,
                    N_, LTOT_, DH_, scale, 0, B_ * HEADS_);

        // masked softmax
        softmax_kernel<<<(unsigned)(B_ * HEADS_ * N_), 256, 0, stream>>>(simbuf);

        // attn_out[b,n,h*96+d] = probs @ vals   (per-head interleave via cDiv)
        launch_gemm(stream, simbuf, LTOT_, (long)N_ * LTOT_,
                    vals, DH_, (long)LTOT_ * DH_, 0,
                    attn_o, H_, (long)N_ * H_, DH_, HEADS_,
                    nullptr, nullptr,
                    N_, DH_, LTOT_, 1.0f, 0, B_ * HEADS_);

        // hidden = hidden + attn_out @ out_W + out_b
        launch_gemm(stream, attn_o, H_, 0,
                    outW + (long)l * H_ * H_, H_, 0, 0,
                    hidden, H_, 0, 0, 1,
                    outb + (long)l * H_, hidden,
                    rows, H_, H_, 1.0f, 0, 1);

        // y = rmsnorm(hidden, norm2)
        rmsnorm_kernel<<<rows, 256, 0, stream>>>(hidden, n2w + (long)l * H_, xnorm);

        // ffmid = gelu(y @ ffn1_W + b)
        launch_gemm(stream, xnorm, H_, 0,
                    f1W + (long)l * H_ * H_, H_, 0, 0,
                    ffmid, H_, 0, 0, 1,
                    f1b + (long)l * H_, nullptr,
                    rows, H_, H_, 1.0f, 1, 1);

        // hidden = hidden + ffmid @ ffn2_W + b
        launch_gemm(stream, ffmid, H_, 0,
                    f2W + (long)l * H_ * H_, H_, 0, 0,
                    hidden, H_, 0, 0, 1,
                    f2b + (long)l * H_, hidden,
                    rows, H_, H_, 1.0f, 0, 1);
    }

    // logits = hidden @ embed^T              [2048,768]x[768,32000]
    launch_gemm(stream, hidden, H_, 0,
                embedW, H_, 0, 1,
                logits, V_, 0, 0, 1,
                nullptr, nullptr,
                rows, V_, H_, 1.0f, 0, 1);
}
